// RouteCapsule_22282290331903
// MI455X (gfx1250) — matrix operand order repaired and checked by reading, not compiled, not run on hardware
//
#include <hip/hip_runtime.h>
#include <hip/hip_bf16.h>

#define O_ 32
#define I_ 1024
#define B_ 64
#define D_ 64
#define E_ 64
#define ICHUNK 32
#define EB_ (E_*B_)   /* 4096 */
#define DE_ (D_*E_)   /* 4096 */

typedef __attribute__((ext_vector_type(16))) __bf16        v16bf;
typedef __attribute__((ext_vector_type(8)))  __bf16        v8bf;
typedef __attribute__((ext_vector_type(8)))  float          v8f;

__device__ __forceinline__ float bf2f(unsigned short h) {
  return __uint_as_float(((unsigned)h) << 16);
}
__device__ __forceinline__ unsigned short bfbits(float f) {
  __bf16 b = (__bf16)f;                 // let backend emit v_cvt_*bf16_f32
  unsigned short s;
  __builtin_memcpy(&s, &b, 2);
  return s;
}

// ---------------------------------------------------------------------------
// K0: x f32 -> bf16, and zero the three s accumulators (contiguous region)
// ---------------------------------------------------------------------------
__global__ __launch_bounds__(256)
void prep_kernel(const float* __restrict__ x, unsigned short* __restrict__ xb,
                 float* __restrict__ szero, int nx, int ns) {
  int idx = blockIdx.x * 256 + threadIdx.x;
  int stride = gridDim.x * 256;
  for (int k = idx; k < nx; k += stride) xb[k] = bfbits(x[k]);
  for (int k = idx; k < ns; k += stride) szero[k] = 0.0f;
}

// ---------------------------------------------------------------------------
// K1: u[o,i,e,b] (bf16) = x[i] @ W[o,i]  via v_wmma_f32_16x16x32_bf16,
//     fused s0[o,e,b] += sum over the block's i-chunk (one atomic pass/block).
// Block = (o, i-chunk of 32), 256 threads = 8 waves.
// Wave w: e-tile et = w&3, b-tiles bt0, bt0+1 with bt0 = 2*(w>>2).
// ---------------------------------------------------------------------------
__global__ __launch_bounds__(256)
void gemm_u_kernel(const unsigned short* __restrict__ xb,
                   const float* __restrict__ W,
                   unsigned short* __restrict__ u,
                   float* __restrict__ s0) {
  const int o     = blockIdx.x & (O_ - 1);
  const int chunk = blockIdx.x >> 5;
  const int wave  = threadIdx.x >> 5;
  const int lane  = threadIdx.x & 31;
  const int lo    = lane & 15;
  const int hi    = lane >> 4;
  const int et    = wave & 3;
  const int bt0   = (wave >> 2) * 2;

  v8f a0 = {}, a1 = {};   // running s0 accumulators (sum over i-chunk)

  for (int ii = 0; ii < ICHUNK; ++ii) {
    const int i = chunk * ICHUNK + ii;
    const float* Wb = W + (size_t)(o * I_ + i) * DE_;
    const unsigned short* x0 = xb + ((size_t)i * B_ + bt0 * 16 + lo) * D_;
    const unsigned short* x1 = x0 + 16 * D_;

    // prefetch next i's W tile (16KB): each thread one 64B region
    if (ii + 1 < ICHUNK)
      __builtin_prefetch(Wb + DE_ + (size_t)threadIdx.x * 16, 0, 1);

    v8f c0 = {}, c1 = {};
#pragma unroll
    for (int s = 0; s < 2; ++s) {
      // A: 16x32 bf16, lane lo = row m, contiguous d runs of 8 at d0 and d0+16
      union { uint4 q[2]; v16bf v; } A0, A1;
      const int d0 = s * 32 + hi * 8;
      A0.q[0] = *(const uint4*)(x0 + d0);
      A0.q[1] = *(const uint4*)(x0 + d0 + 16);
      A1.q[0] = *(const uint4*)(x1 + d0);
      A1.q[1] = *(const uint4*)(x1 + d0 + 16);
      // B: 32x16, element j -> k = s*32 + hi*8 + j + (j>=8 ? 8 : 0), n = et*16 + lo
      // per-j, lanes 0..15 read 64B contiguous -> coalesced
      v16bf Bv;
      const float* wp = Wb + (size_t)(s * 32 + hi * 8) * E_ + et * 16 + lo;
#pragma unroll
      for (int j = 0; j < 16; ++j) Bv[j] = (__bf16)wp[(size_t)(j + (j >= 8 ? 8 : 0)) * E_];

      c0 = __builtin_amdgcn_wmma_f32_16x16x32_bf16(false, A0.v, false, Bv,
                                                   (short)0, c0, false, false);
      c1 = __builtin_amdgcn_wmma_f32_16x16x32_bf16(false, A1.v, false, Bv,
                                                   (short)0, c1, false, false);
    }
#pragma unroll
    for (int r = 0; r < 8; ++r) { a0[r] += c0[r]; a1[r] += c1[r]; }

    // store u tiles: D layout lane(lo)=e col, 8 consecutive b -> one b128 each
    unsigned short* ub = u + (size_t)(o * I_ + i) * EB_ + (et * 16 + lo) * B_;
    union { v8bf v; uint4 q; } P0, P1;
#pragma unroll
    for (int r = 0; r < 8; ++r) { P0.v[r] = (__bf16)c0[r]; P1.v[r] = (__bf16)c1[r]; }
    *(uint4*)(ub + bt0 * 16 + hi * 8)       = P0.q;
    *(uint4*)(ub + (bt0 + 1) * 16 + hi * 8) = P1.q;
  }

  float* sb = s0 + (size_t)o * EB_ + (et * 16 + lo) * B_;
#pragma unroll
  for (int r = 0; r < 8; ++r) {
    atomicAdd(&sb[bt0 * 16 + hi * 8 + r], a0[r]);
    atomicAdd(&sb[(bt0 + 1) * 16 + hi * 8 + r], a1[r]);
  }
}

// ---------------------------------------------------------------------------
// K2: squash. v[o,e,b] = squash_e(s[o,:,b]*scale); optionally write d_out
//     in reference [O,B,E] layout. 2048 threads (one per (o,b)).
// ---------------------------------------------------------------------------
__global__ __launch_bounds__(256)
void squash_kernel(const float* __restrict__ s, float scale,
                   float* __restrict__ vint, float* __restrict__ out_obe) {
  int tid = blockIdx.x * 256 + threadIdx.x;   // 0..2047
  int o = tid >> 6, b = tid & 63;
  const float* sb = s + (size_t)o * EB_ + b;
  float sq = 0.0f;
#pragma unroll
  for (int e = 0; e < E_; ++e) { float t = sb[e * B_] * scale; sq += t * t; }
  float f = sq / (1.0f + sq) * rsqrtf(sq);
#pragma unroll
  for (int e = 0; e < E_; ++e) {
    float val = sb[e * B_] * scale * f;
    vint[(size_t)o * EB_ + e * B_ + b] = val;
    if (out_obe) out_obe[(size_t)(o * B_ + b) * E_ + e] = val;
  }
}

// ---------------------------------------------------------------------------
// K3: agreement. bnew[o,i,b] = (bold?bold:0) + sum_e u[o,i,e,b]*v[o,e,b]
// Block = (o, i-chunk of 16); v[o] staged in LDS once per block.
// ---------------------------------------------------------------------------
__global__ __launch_bounds__(256)
void agree_kernel(const unsigned short* __restrict__ u,
                  const float* __restrict__ v,
                  const float* __restrict__ bold, float* __restrict__ bnew) {
  __shared__ float lv[EB_];
  __shared__ float red[256];
  const int o = blockIdx.x >> 6;
  const int chunk = blockIdx.x & 63;
  const int t = threadIdx.x, b = t & 63, g = t >> 6;
  const float* vb = v + (size_t)o * EB_;
  for (int k = t; k < EB_; k += 256) lv[k] = vb[k];
  __syncthreads();
  for (int ii = 0; ii < 16; ++ii) {
    const int i = chunk * 16 + ii;
    const unsigned short* ub = u + (size_t)(o * I_ + i) * EB_;
    // prefetch next i's u slab (8KB): threads 0..63 cover it in 128B lines
    if (ii + 1 < 16 && t < 64) __builtin_prefetch(ub + EB_ + t * 64, 0, 1);
    float p = 0.0f;
#pragma unroll
    for (int e = g * 16; e < g * 16 + 16; ++e) {
      int idx = e * B_ + b;
      p += bf2f(ub[idx]) * lv[idx];
    }
    red[t] = p;
    __syncthreads();
    if (g == 0) {
      float tot = red[b] + red[64 + b] + red[128 + b] + red[192 + b];
      size_t bi = (size_t)(o * I_ + i) * B_ + b;
      bnew[bi] = (bold ? bold[bi] : 0.0f) + tot;
    }
    __syncthreads();
  }
}

// ---------------------------------------------------------------------------
// K4: softmax over o for each (i,b). 65536 threads.
// ---------------------------------------------------------------------------
__global__ __launch_bounds__(256)
void softmax_kernel(const float* __restrict__ bb, float* __restrict__ c) {
  int tid = blockIdx.x * 256 + threadIdx.x;  // 0..I_*B_-1
  float vals[O_];
  float m = -1e30f;
#pragma unroll
  for (int o = 0; o < O_; ++o) {
    vals[o] = bb[(size_t)o * (I_ * B_) + tid];
    m = fmaxf(m, vals[o]);
  }
  float ssum = 0.0f;
#pragma unroll
  for (int o = 0; o < O_; ++o) { vals[o] = __expf(vals[o] - m); ssum += vals[o]; }
  float inv = 1.0f / ssum;
#pragma unroll
  for (int o = 0; o < O_; ++o) c[(size_t)o * (I_ * B_) + tid] = vals[o] * inv;
}

// ---------------------------------------------------------------------------
// K5: weighted sum. s[o,e,b] += sum_i c[o,i,b]*u[o,i,e,b].
// Block = (o, i-chunk of 16). Each thread owns fixed (e,b) cells across the
// whole i-loop -> accumulate in 16 registers, then one atomic pass.
// ---------------------------------------------------------------------------
__global__ __launch_bounds__(256)
void wsum_kernel(const unsigned short* __restrict__ u,
                 const float* __restrict__ c, float* __restrict__ s) {
  const int o = blockIdx.x >> 6;
  const int chunk = blockIdx.x & 63;
  const int t = threadIdx.x, b = t & 63, e0 = t >> 6;
  float acc[16];
#pragma unroll
  for (int k = 0; k < 16; ++k) acc[k] = 0.0f;
  for (int ii = 0; ii < 16; ++ii) {
    const int i = chunk * 16 + ii;
    const float cb = c[(size_t)(o * I_ + i) * B_ + b];
    const unsigned short* ub = u + (size_t)(o * I_ + i) * EB_;
    if (ii + 1 < 16 && t < 64) __builtin_prefetch(ub + EB_ + t * 64, 0, 1);
#pragma unroll
    for (int k = 0; k < 16; ++k) {
      int idx = (e0 + 4 * k) * B_ + b;
      acc[k] += cb * bf2f(ub[idx]);
    }
  }
  float* sb = s + (size_t)o * EB_;
#pragma unroll
  for (int k = 0; k < 16; ++k)
    atomicAdd(&sb[(e0 + 4 * k) * B_ + b], acc[k]);
}

// ---------------------------------------------------------------------------
extern "C" void kernel_launch(void* const* d_in, const int* in_sizes, int n_in,
                              void* d_out, int out_size, void* d_ws, size_t ws_size,
                              hipStream_t stream) {
  const float* x = (const float*)d_in[0];        // [I,B,D] f32
  const float* W = (const float*)d_in[1];        // [O,I,D,E] f32
  float* out = (float*)d_out;                    // [O,B,E] f32

  char* ws = (char*)d_ws;
  size_t off = 0;
  unsigned short* xb = (unsigned short*)(ws + off); off += (size_t)I_ * B_ * D_ * 2;
  unsigned short* u  = (unsigned short*)(ws + off); off += (size_t)O_ * I_ * EB_ * 2;
  float* s0 = (float*)(ws + off); off += (size_t)O_ * EB_ * 4;
  float* s1 = (float*)(ws + off); off += (size_t)O_ * EB_ * 4;
  float* s2 = (float*)(ws + off); off += (size_t)O_ * EB_ * 4;
  float* v0 = (float*)(ws + off); off += (size_t)O_ * EB_ * 4;
  float* v1 = (float*)(ws + off); off += (size_t)O_ * EB_ * 4;
  float* b1 = (float*)(ws + off); off += (size_t)O_ * I_ * B_ * 4;
  float* b2 = (float*)(ws + off); off += (size_t)O_ * I_ * B_ * 4;
  float* cc = (float*)(ws + off); off += (size_t)O_ * I_ * B_ * 4;

  // s0,s1,s2 are contiguous -> zero them in one grid-stride pass
  prep_kernel<<<4096, 256, 0, stream>>>(x, xb, s0, I_ * B_ * D_, 3 * O_ * EB_);

  // u + fused s0 = sum_i u
  gemm_u_kernel<<<O_ * (I_ / ICHUNK), 256, 0, stream>>>(xb, W, u, s0);

  // it 0: c uniform = 1/32 -> v0 = squash(s0/32)
  squash_kernel<<<8, 256, 0, stream>>>(s0, 1.0f / 32.0f, v0, nullptr);

  // it 1
  agree_kernel<<<2048, 256, 0, stream>>>(u, v0, nullptr, b1);
  softmax_kernel<<<256, 256, 0, stream>>>(b1, cc);
  wsum_kernel<<<2048, 256, 0, stream>>>(u, cc, s1);
  squash_kernel<<<8, 256, 0, stream>>>(s1, 1.0f, v1, nullptr);

  // it 2 (final)
  agree_kernel<<<2048, 256, 0, stream>>>(u, v1, b1, b2);
  softmax_kernel<<<256, 256, 0, stream>>>(b2, cc);
  wsum_kernel<<<2048, 256, 0, stream>>>(u, cc, s2);
  squash_kernel<<<8, 256, 0, stream>>>(s2, 1.0f, v0, out);
}
